// MixedSquarePatchEmbed_20486994002012
// MI455X (gfx1250) — matrix operand order repaired and checked by reading, not compiled, not run on hardware
//
#include <hip/hip_runtime.h>
#include <hip/hip_bf16.h>

typedef _Float16 half_t;
typedef __attribute__((ext_vector_type(8)))  _Float16 v8h;
typedef __attribute__((ext_vector_type(16))) _Float16 v16h;
typedef __attribute__((ext_vector_type(8)))  float    v8f;
typedef __attribute__((ext_vector_type(4)))  float    v4f;

// Problem dims
constexpr int Bb = 64, Cc = 3, Hh = 384, Pp = 16, Dd = 768;
constexpr int hh = 24, hcc = 12, DHh = 192;
constexpr int NPATCH = hh * hh;        // 576
constexpr int NGRP   = hcc * hcc;      // 144
constexpr int MTOK   = Bb * NPATCH;    // 36864
constexpr int MGRP   = Bb * NGRP;      // 9216
constexpr int OUTROW = NPATCH + NGRP;  // 720
constexpr float THR_V = 6.0f;

// -------------------- weight transpose + f32->f16 convert --------------------
// WT[n][k] = (f16) W[k][nofs + n]
__global__ void wt_cvt_kernel(const float* __restrict__ W, half_t* __restrict__ WT,
                              int K, int Nout, int nofs, int Nsrc) {
  int id = blockIdx.x * 256 + threadIdx.x;
  if (id >= K * Nout) return;
  int k = id / Nout, n = id % Nout;
  WT[(size_t)n * K + k] = (half_t)W[(size_t)k * Nsrc + nofs + n];
}

// -------------------- patch gather: x -> f16 A matrix ------------------------
// A[row][c*256+py*16+px] = x[b, c, gy*16+py, gx*16+px], row = b*576+gy*24+gx
__global__ void patch_gather_kernel(const float* __restrict__ x, half_t* __restrict__ Ap) {
  int id = blockIdx.x * 256 + threadIdx.x;       // MTOK * 48 chunk ids
  if (id >= MTOK * 48) return;
  int row = id / 48, ch = id % 48;
  int c = ch >> 4, py = ch & 15;
  int b = row / NPATCH, p = row % NPATCH;
  int gy = p / hh, gx = p % hh;
  const float* src = x + ((size_t)((b * Cc + c) * Hh) + gy * Pp + py) * Hh + gx * Pp;
  half_t* dst = Ap + (size_t)row * Dd + ch * 16;
  v4f f0 = *(const v4f*)(src + 0);
  v4f f1 = *(const v4f*)(src + 4);
  v4f f2 = *(const v4f*)(src + 8);
  v4f f3 = *(const v4f*)(src + 12);
  v8h h0, h1;
#pragma unroll
  for (int i = 0; i < 4; ++i) {
    h0[i]     = (_Float16)f0[i];
    h0[i + 4] = (_Float16)f1[i];
    h1[i]     = (_Float16)f2[i];
    h1[i + 4] = (_Float16)f3[i];
  }
  *(v8h*)(dst)     = h0;
  *(v8h*)(dst + 8) = h1;
}

// -------------------- constant qkv for the shared group token ----------------
__global__ void qkv0_kernel(const float* __restrict__ Wqkv, const float* __restrict__ bqkv,
                            const float* __restrict__ gtok, float* __restrict__ qkv0) {
  int j = blockIdx.x * 256 + threadIdx.x;
  if (j >= 3 * Dd) return;
  float acc = bqkv[j];
  for (int k = 0; k < Dd; ++k) acc += gtok[k] * Wqkv[(size_t)k * (3 * Dd) + j];
  qkv0[j] = acc;
}

// -------------------- CDNA5 async global->LDS copy (32 bytes / thread) -------
// GLOBAL_LOAD_ASYNC_TO_LDS_B128: per-lane 16B, tracked by ASYNCcnt, writes LDS
// directly. The instruction offset is added to BOTH the LDS and global address
// (CDNA5 ISA), so one (lds, global) pair covers both chunks.
__device__ __forceinline__ void async_ld32B(unsigned lds_byte, unsigned long long gaddr) {
  asm volatile(
      "global_load_async_to_lds_b128 %0, %1, off\n\t"
      "global_load_async_to_lds_b128 %0, %1, off offset:16"
      :: "v"(lds_byte), "v"(gaddr) : "memory");
}

// -------------------- f16 NT GEMM with WMMA, fused epilogues -----------------
// C[m][n] = sum_k A[m][k] * Bt[n][k]   (A: M x K row-major, Bt: N x K row-major)
// Block tile 128x128, K-step 32, 256 threads = 8 waves in a 2(m) x 4(n) grid,
// each wave does 4x2 frags of v_wmma_f32_16x16x32_f16. Double-buffered LDS fed
// by global_load_async_to_lds_b128; overlap via s_wait_asynccnt.
union AFrag { v16h v; v8h h[2]; };

template <int EPI>
__global__ __launch_bounds__(256) void gemm_nt_kernel(
    const half_t* __restrict__ A, const half_t* __restrict__ Bt, int K,
    const float* __restrict__ bias, const float* __restrict__ pos,
    const float* __restrict__ ent, float* __restrict__ outf, half_t* __restrict__ outh) {
  constexpr int BK = 32, LDS_S = BK + 8;          // stride 40 halves: 16B align kept
  constexpr int BUFH = 128 * LDS_S;               // halves per buffer
  constexpr unsigned BUFBYTES = BUFH * 2;         // 10240 B
  __shared__ __align__(16) half_t As[2 * BUFH];   // 20 KB
  __shared__ __align__(16) half_t Bs[2 * BUFH];   // 20 KB

  const int tid  = threadIdx.x;
  const int lane = tid & 31;
  const int wave = tid >> 5;
  const int wm = wave >> 2, wn = wave & 3;
  const int m0 = blockIdx.y * 128, n0 = blockIdx.x * 128;

  // async staging: thread covers one 32B run (16 halves) of one row per tile
  const int lrow = tid >> 1;
  const int lkb  = (tid & 1) * 16;                // halves
  const unsigned ldsRel = (unsigned)((lrow * LDS_S + lkb) * 2);
  const unsigned asBase = (unsigned)(uintptr_t)(&As[0]);  // LDS aperture: low 32b = LDS offset
  const unsigned bsBase = (unsigned)(uintptr_t)(&Bs[0]);
  const half_t* gA = A  + (size_t)(m0 + lrow) * K + lkb;
  const half_t* gB = Bt + (size_t)(n0 + lrow) * K + lkb;

  v8f acc[4][2];
#pragma unroll
  for (int i = 0; i < 4; ++i)
#pragma unroll
    for (int j = 0; j < 2; ++j)
#pragma unroll
      for (int r = 0; r < 8; ++r) acc[i][j][r] = 0.0f;

  // prologue: tile 0 -> buffer 0
  async_ld32B(asBase + ldsRel, (unsigned long long)(uintptr_t)gA);
  async_ld32B(bsBase + ldsRel, (unsigned long long)(uintptr_t)gB);

  int buf = 0;
  for (int kt = 0; kt < K; kt += BK, buf ^= 1) {
    if (kt + BK < K) {
      // issue tile kt+BK into the other buffer, then wait for current tile's
      // 4 async ops (leave the 4 newest in flight).
      const unsigned bo = (unsigned)(buf ^ 1) * BUFBYTES;
      async_ld32B(asBase + bo + ldsRel, (unsigned long long)(uintptr_t)(gA + kt + BK));
      async_ld32B(bsBase + bo + ldsRel, (unsigned long long)(uintptr_t)(gB + kt + BK));
      if (kt + 2 * BK < K) {  // warm L2 for tile t+2 (global_prefetch_b8)
        __builtin_prefetch(gA + kt + 2 * BK, 0, 1);
        __builtin_prefetch(gB + kt + 2 * BK, 0, 1);
      }
      asm volatile("s_wait_asynccnt 0x4" ::: "memory");
    } else {
      asm volatile("s_wait_asynccnt 0x0" ::: "memory");
    }
    __syncthreads();  // current buffer visible to all waves

    const half_t* cA = &As[buf * BUFH];
    const half_t* cB = &Bs[buf * BUFH];
    AFrag af[4], bf[2];
    // A 16x32 f16 frag: lanes 0-15 K={0..7,16..23}, lanes 16-31 K={8..15,24..31}
    const int akb = (lane & 16) ? 8 : 0;
#pragma unroll
    for (int mf = 0; mf < 4; ++mf) {
      const half_t* p = &cA[(wm * 64 + mf * 16 + (lane & 15)) * LDS_S + akb];
      af[mf].h[0] = *(const v8h*)p;
      af[mf].h[1] = *(const v8h*)(p + 16);
    }
    // B 32x16 f16 frag: lanes 0-15 K={0..7,16..23}, lanes 16-31 K={8..15,24..31} (col n per lane)
    const int bkb = (lane & 16) ? 8 : 0;
#pragma unroll
    for (int nf = 0; nf < 2; ++nf) {
      const half_t* p = &cB[(wn * 32 + nf * 16 + (lane & 15)) * LDS_S + bkb];
      bf[nf].h[0] = *(const v8h*)p;
      bf[nf].h[1] = *(const v8h*)(p + 16);
    }
#pragma unroll
    for (int mf = 0; mf < 4; ++mf)
#pragma unroll
      for (int nf = 0; nf < 2; ++nf)
        acc[mf][nf] = __builtin_amdgcn_wmma_f32_16x16x32_f16(
            false, af[mf].v, false, bf[nf].v, (short)0, acc[mf][nf], false, false);

    __syncthreads();  // all waves done reading before this buffer is re-filled
  }

  // Epilogue: C/D layout: lane n=lane&15, rows m = (lane>=16 ? 8 : 0) + reg
  const int msel = (lane >> 4) & 1, nid = lane & 15;
#pragma unroll
  for (int mf = 0; mf < 4; ++mf) {
#pragma unroll
    for (int nf = 0; nf < 2; ++nf) {
      const int ncol = n0 + wn * 32 + nf * 16 + nid;
#pragma unroll
      for (int r = 0; r < 8; ++r) {
        const int mrow = m0 + wm * 64 + mf * 16 + msel * 8 + r;
        float v = acc[mf][nf][r];
        if (EPI == 0) {
          // patch embed: + b_patch + pos_embed; write f16 tok + fine-masked f32 out
          int b = mrow / NPATCH, p = mrow % NPATCH;
          v += bias[ncol] + pos[(size_t)p * Dd + ncol];
          int gy = p / hh, gx = p % hh;
          float e = ent[b * NGRP + (gy >> 1) * hcc + (gx >> 1)];
          outh[(size_t)mrow * Dd + ncol] = (half_t)v;
          outf[((size_t)b * OUTROW + p) * Dd + ncol] = (e >= THR_V) ? v : 0.0f;
        } else if (EPI == 1) {
          // kv: + b_qkv[768+ncol] (bias pre-offset); write f16 KV buffer
          v += bias[ncol];
          outh[(size_t)mrow * (2 * Dd) + ncol] = (half_t)v;
        } else {
          // proj: + b_proj; coarse-masked f32 out
          v += bias[ncol];
          int b = mrow / NGRP, g = mrow % NGRP;
          float e = ent[mrow];
          outf[((size_t)b * OUTROW + NPATCH + g) * Dd + ncol] = (e < THR_V) ? v : 0.0f;
        }
      }
    }
  }
}

// -------------------- per-group 5-token attention (group-token row only) -----
__global__ __launch_bounds__(768) void attn_kernel(const float* __restrict__ qkv0,
                                                   const half_t* __restrict__ kv,
                                                   half_t* __restrict__ ctx) {
  __shared__ float red[24][5];
  const int gid = blockIdx.x;  // b*144 + g
  const int b = gid / NGRP, g = gid % NGRP;
  const int cy = g / hcc, cx = g % hcc;
  const int tid = threadIdx.x;           // feature dim d in [0,768)
  const int lane = tid & 31, wave = tid >> 5;
  const int h = tid / DHh;               // head (0..3), 6 waves per head

  int rows[4];
#pragma unroll
  for (int j = 0; j < 4; ++j) {
    int dy = j >> 1, dx = j & 1;
    rows[j] = b * NPATCH + (2 * cy + dy) * hh + (2 * cx + dx);
  }
  const float q = qkv0[tid];             // q0 (constant group token query)
  float part[5], vvec[5];
  part[0] = q * qkv0[Dd + tid];          // k0
  vvec[0] = qkv0[2 * Dd + tid];          // v0
#pragma unroll
  for (int j = 0; j < 4; ++j) {
    const half_t* kr = kv + (size_t)rows[j] * (2 * Dd);
    part[j + 1] = q * (float)kr[tid];
    vvec[j + 1] = (float)kr[Dd + tid];
  }
#pragma unroll
  for (int j = 0; j < 5; ++j)
    for (int off = 16; off > 0; off >>= 1) part[j] += __shfl_down(part[j], off, 32);
  if (lane == 0)
#pragma unroll
    for (int j = 0; j < 5; ++j) red[wave][j] = part[j];
  __syncthreads();

  const float scale = 0.07216878364870323f;  // 1/sqrt(192)
  float l[5];
#pragma unroll
  for (int j = 0; j < 5; ++j) {
    float s = 0.0f;
#pragma unroll
    for (int w = 0; w < 6; ++w) s += red[h * 6 + w][j];
    l[j] = s * scale;
  }
  float mx = l[0];
#pragma unroll
  for (int j = 1; j < 5; ++j) mx = fmaxf(mx, l[j]);
  float se = 0.0f, p[5];
#pragma unroll
  for (int j = 0; j < 5; ++j) { p[j] = __expf(l[j] - mx); se += p[j]; }
  const float inv = 1.0f / se;
  float o = 0.0f;
#pragma unroll
  for (int j = 0; j < 5; ++j) o += p[j] * inv * vvec[j];
  ctx[(size_t)gid * Dd + tid] = (half_t)o;
}

// -----------------------------------------------------------------------------
extern "C" void kernel_launch(void* const* d_in, const int* in_sizes, int n_in,
                              void* d_out, int out_size, void* d_ws, size_t ws_size,
                              hipStream_t stream) {
  (void)in_sizes; (void)n_in; (void)out_size; (void)ws_size;
  const float* x   = (const float*)d_in[0];
  const float* ent = (const float*)d_in[1];
  const float* Wp  = (const float*)d_in[2];
  const float* bp  = (const float*)d_in[3];
  const float* pos = (const float*)d_in[4];
  const float* gt  = (const float*)d_in[5];
  const float* Wq  = (const float*)d_in[6];
  const float* bq  = (const float*)d_in[7];
  const float* Wo  = (const float*)d_in[8];
  const float* bo  = (const float*)d_in[9];
  float* out = (float*)d_out;

  char* ws = (char*)d_ws;
  // workspace layout (bytes), all 256B aligned
  float*  qkv0 = (float*)(ws + 0);                        //   9,216
  half_t* WTp  = (half_t*)(ws + 9216);                    // 1,179,648
  half_t* WTkv = (half_t*)(ws + 1188864);                 // 2,359,296
  half_t* WTo  = (half_t*)(ws + 3548160);                 // 1,179,648
  half_t* Ap   = (half_t*)(ws + 4727808);                 // 56,623,104
  half_t* tok  = (half_t*)(ws + 61350912);                // 56,623,104
  half_t* kvb  = (half_t*)(ws + 117974016);               // 113,246,208
  half_t* ctx  = (half_t*)(ws + 231220224);               // 14,155,776  (total ~245 MB)

  // 1) weights -> f16, transposed to [n][k]
  wt_cvt_kernel<<<(Dd * Dd + 255) / 256, 256, 0, stream>>>(Wp, WTp, Dd, Dd, 0, Dd);
  wt_cvt_kernel<<<(Dd * 2 * Dd + 255) / 256, 256, 0, stream>>>(Wq, WTkv, Dd, 2 * Dd, Dd, 3 * Dd);
  wt_cvt_kernel<<<(Dd * Dd + 255) / 256, 256, 0, stream>>>(Wo, WTo, Dd, Dd, 0, Dd);

  // 2) patch gather x -> f16 A
  patch_gather_kernel<<<(MTOK * 48 + 255) / 256, 256, 0, stream>>>(x, Ap);

  // 3) constant group-token qkv
  qkv0_kernel<<<(3 * Dd + 255) / 256, 256, 0, stream>>>(Wq, bq, gt, qkv0);

  // 4) GEMM1: tok = A @ W_patch (+b+pos); fused fine-mask output + f16 tok
  gemm_nt_kernel<0><<<dim3(Dd / 128, MTOK / 128), 256, 0, stream>>>(
      Ap, WTp, Dd, bp, pos, ent, out, tok);

  // 5) GEMM2: kv = tok @ W_qkv[:,768:2304] (+b)
  gemm_nt_kernel<1><<<dim3(2 * Dd / 128, MTOK / 128), 256, 0, stream>>>(
      tok, WTkv, Dd, bq + Dd, nullptr, nullptr, nullptr, kvb);

  // 6) attention (group-token row only)
  attn_kernel<<<MGRP, 768, 0, stream>>>(qkv0, kvb, ctx);

  // 7) GEMM3: merged = ctx @ W_proj (+b); fused coarse-mask output
  gemm_nt_kernel<2><<<dim3(Dd / 128, MGRP / 128), 256, 0, stream>>>(
      ctx, WTo, Dd, bo, nullptr, ent, out, nullptr);
}
